// RWKV_67284957659398
// MI455X (gfx1250) — matrix operand order repaired and checked by reading, not compiled, not run on hardware
//
#include <hip/hip_runtime.h>

// ---------------------------------------------------------------------------
// RWKV block for MI455X (gfx1250): bf16 WMMA GEMMs with async global->LDS
// double-buffered staging + fp32 WKV scan fused with the r-gate.
// ---------------------------------------------------------------------------

#define B_SZ   8
#define T_SZ   2048
#define C_SZ   1024
#define FFN_SZ 4096
#define BT_SZ  (B_SZ * T_SZ)          // 16384 tokens
#define EPS_LN 1e-5f

typedef __attribute__((ext_vector_type(16))) __bf16 bf16x16;
typedef __attribute__((ext_vector_type(8)))  __bf16 bf16x8;
typedef __attribute__((ext_vector_type(8)))  float  f32x8;

union FragBF16 { bf16x16 v; bf16x8 h[2]; };

enum { ACT_NONE = 0, ACT_SIGMOID = 1, ACT_RELUSQ = 2 };

// ---------------------------------------------------------------------------
// Async 16B global -> LDS copy (CDNA5 GLOBAL_LOAD_ASYNC_TO_LDS_B128).
// LDS dest VGPR holds the LDS byte offset; generic LDS pointers carry the
// offset in their low 32 bits.
// ---------------------------------------------------------------------------
__device__ __forceinline__ void async_copy_b128(const __bf16* gsrc,
                                                const __bf16* ldst) {
  const unsigned lds_off = (unsigned)(unsigned long long)(const void*)ldst;
  asm volatile("global_load_async_to_lds_b128 %0, %1, off"
               :: "v"(lds_off), "v"(gsrc) : "memory");
}

__device__ __forceinline__ void wait_async_le4() {
  asm volatile("s_wait_asynccnt 0x4" ::: "memory");
}
__device__ __forceinline__ void wait_async_0() {
  asm volatile("s_wait_asynccnt 0x0" ::: "memory");
}

// ---------------------------------------------------------------------------
// Weight transpose + fp32 -> bf16 convert: in[K,N] row-major -> out[N,K].
// ---------------------------------------------------------------------------
__global__ __launch_bounds__(256) void transpose_to_bf16(
    const float* __restrict__ in, __bf16* __restrict__ out, int K, int N) {
  __shared__ float tile[32][33];
  const int n0 = blockIdx.x * 32;
  const int k0 = blockIdx.y * 32;
  const int tx = threadIdx.x;      // 0..31
  const int ty = threadIdx.y;      // 0..7
#pragma unroll
  for (int j = 0; j < 32; j += 8)
    tile[ty + j][tx] = in[(size_t)(k0 + ty + j) * N + (n0 + tx)];
  __syncthreads();
#pragma unroll
  for (int j = 0; j < 32; j += 8)
    out[(size_t)(n0 + ty + j) * K + (k0 + tx)] = (__bf16)tile[tx][ty + j];
}

// ---------------------------------------------------------------------------
// LayerNorm over last dim (C=1024). One 256-thread block per token row.
// ---------------------------------------------------------------------------
__global__ __launch_bounds__(256) void ln_kernel(
    const float* __restrict__ x, const float* __restrict__ w,
    const float* __restrict__ b, float* __restrict__ y) {
  __shared__ float red[256];
  const int row = blockIdx.x;
  const int t   = threadIdx.x;
  const float* xr = x + (size_t)row * C_SZ;

  float v[4];
  float s = 0.f;
#pragma unroll
  for (int i = 0; i < 4; ++i) { v[i] = xr[t + 256 * i]; s += v[i]; }
  red[t] = s;
  __syncthreads();
  for (int off = 128; off > 0; off >>= 1) {
    if (t < off) red[t] += red[t + off];
    __syncthreads();
  }
  const float mean = red[0] * (1.0f / C_SZ);
  __syncthreads();

  float s2 = 0.f;
#pragma unroll
  for (int i = 0; i < 4; ++i) { float d = v[i] - mean; s2 += d * d; }
  red[t] = s2;
  __syncthreads();
  for (int off = 128; off > 0; off >>= 1) {
    if (t < off) red[t] += red[t + off];
    __syncthreads();
  }
  const float rinv = rsqrtf(red[0] * (1.0f / C_SZ) + EPS_LN);

  float* yr = y + (size_t)row * C_SZ;
#pragma unroll
  for (int i = 0; i < 4; ++i) {
    const int c = t + 256 * i;
    yr[c] = (v[i] - mean) * rinv * w[c] + b[c];
  }
}

// ---------------------------------------------------------------------------
// Token-shift mixing: out = xs + mix*(xn - xs), xs = xn[t-1] (0 at t==0).
// ---------------------------------------------------------------------------
__global__ __launch_bounds__(256) void mix3_kernel(
    const float* __restrict__ xn, const float* __restrict__ mk,
    const float* __restrict__ mv, const float* __restrict__ mr,
    __bf16* __restrict__ kx, __bf16* __restrict__ vx, __bf16* __restrict__ rx) {
  const size_t idx = (size_t)blockIdx.x * 256 + threadIdx.x;
  const int c = (int)(idx & (C_SZ - 1));
  const size_t bt = idx >> 10;
  const int tt = (int)(bt & (T_SZ - 1));
  const float xv = xn[idx];
  const float xs = (tt == 0) ? 0.f : xn[idx - C_SZ];
  const float d = xv - xs;
  kx[idx] = (__bf16)(xs + mk[c] * d);
  vx[idx] = (__bf16)(xs + mv[c] * d);
  rx[idx] = (__bf16)(xs + mr[c] * d);
}

__global__ __launch_bounds__(256) void mix2_kernel(
    const float* __restrict__ yn, const float* __restrict__ mk,
    const float* __restrict__ mr,
    __bf16* __restrict__ kx, __bf16* __restrict__ rx) {
  const size_t idx = (size_t)blockIdx.x * 256 + threadIdx.x;
  const int c = (int)(idx & (C_SZ - 1));
  const size_t bt = idx >> 10;
  const int tt = (int)(bt & (T_SZ - 1));
  const float xv = yn[idx];
  const float xs = (tt == 0) ? 0.f : yn[idx - C_SZ];
  const float d = xv - xs;
  kx[idx] = (__bf16)(xs + mk[c] * d);
  rx[idx] = (__bf16)(xs + mr[c] * d);
}

// ---------------------------------------------------------------------------
// Numerically-stabilized WKV recurrence fused with the sigmoid(r) gate:
// writes bf16(r * wkv) ready for the Wo GEMM. One thread per (b, c) channel.
// ---------------------------------------------------------------------------
__global__ __launch_bounds__(256) void wkv_gate_kernel(
    const float* __restrict__ k, const float* __restrict__ v,
    const float* __restrict__ r,
    const float* __restrict__ time_first, const float* __restrict__ time_decay,
    __bf16* __restrict__ gated) {
  const int i = blockIdx.x * 256 + threadIdx.x;   // 0 .. B*C-1
  const int b = i / C_SZ;
  const int c = i & (C_SZ - 1);
  const float u = time_first[c];
  const float w = -__expf(time_decay[c]);

  const size_t base = (size_t)b * T_SZ * C_SZ + c;
  const float* kp = k + base;
  const float* vp = v + base;
  const float* rp = r + base;
  __bf16* op = gated + base;

  float aa = 0.f, bb = 0.f, pp = 1.0f - 1e30f;
  for (int t = 0; t < T_SZ; ++t) {
    const size_t off = (size_t)t * C_SZ;
    const float kt = kp[off];
    const float vt = vp[off];
    const float uk = kt + u;
    const float q  = fmaxf(pp, uk);
    const float e1 = __expf(pp - q);
    const float e2 = __expf(uk - q);
    const float num = e1 * aa + e2 * vt;
    const float den = e1 * bb + e2;
    const float wkv = (den == 0.f) ? 0.f : num / den;
    op[off] = (__bf16)(rp[off] * wkv);
    const float cw = w + pp;
    const float q2 = fmaxf(cw, kt);
    const float f1 = __expf(cw - q2);
    const float f2 = __expf(kt - q2);
    aa = f1 * aa + f2 * vt;
    bb = f1 * bb + f2;
    pp = q2;
  }
}

// ---------------------------------------------------------------------------
// bf16 WMMA GEMM: C[M,N] = act(A[M,K] * Bt[N,K]^T) [* gate] [+ residual].
//   A row-major bf16; Bt = transposed weights, row-major [N,K] bf16.
//   Block tile 128x128, BK=32, 256 threads (8 wave32), wave tile 32x64.
//   Global -> LDS via async b128 copies, double-buffered (ASYNCcnt pipeline);
//   last stage peeled so the steady-state loop body is branch-free.
// ---------------------------------------------------------------------------
template <int ACT, typename OutT>
__global__ __launch_bounds__(256) void gemm_bf16_wmma(
    const __bf16* __restrict__ A, const __bf16* __restrict__ Bt,
    OutT* __restrict__ Cmat, const float* __restrict__ residual,
    const float* __restrict__ gatemul, int M, int N, int K) {
  constexpr int BM = 128, BN = 128, BK = 32, LDSP = 40;  // pad 32->40 halves
  __shared__ __bf16 lA[2][BM * LDSP];
  __shared__ __bf16 lB[2][BN * LDSP];

  const int tid  = threadIdx.x;
  const int lane = tid & 31;
  const int wave = tid >> 5;
  const int wm   = (wave & 3) * 32;   // wave M offset within block tile
  const int wn   = (wave >> 2) * 64;  // wave N offset within block tile
  const int m0   = blockIdx.y * BM;
  const int n0   = blockIdx.x * BN;

  const int lrow = tid >> 2;          // 0..63 (+64 for second pass)
  const int lcol = (tid & 3) * 8;     // halves within BK
  const int lo   = lrow * LDSP + lcol;

  // Per-thread global source pointers; advanced incrementally by BK.
  const __bf16* pA0 = A  + (size_t)(m0 + lrow)      * K + lcol;
  const __bf16* pA1 = A  + (size_t)(m0 + lrow + 64) * K + lcol;
  const __bf16* pB0 = Bt + (size_t)(n0 + lrow)      * K + lcol;
  const __bf16* pB1 = Bt + (size_t)(n0 + lrow + 64) * K + lcol;

  f32x8 acc[2][4] = {};

  // Fragment indexing per CDNA5 ISA §7.12.2 (16-bit A 16x32, B 32x16):
  const int am  = lane & 15;
  const int akb = (lane & 16) ? 8 : 0;    // A: K-base 0 / 8 (plus +16 chunk)
  const int bkb = (lane & 16) ? 8 : 0;    // B: K-base 0 / 8 (plus +16 chunk)
  const int bn  = lane & 15;

  // 12 ds_load_b128 + 8 v_wmma per stage.
  auto compute_stage = [&](const __bf16* cA, const __bf16* cB) {
    FragBF16 afr[2];
#pragma unroll
    for (int tm = 0; tm < 2; ++tm) {
      const int row = wm + tm * 16 + am;
      afr[tm].h[0] = *(const bf16x8*)(cA + row * LDSP + akb);
      afr[tm].h[1] = *(const bf16x8*)(cA + row * LDSP + akb + 16);
    }
    FragBF16 bfr[4];
#pragma unroll
    for (int tn = 0; tn < 4; ++tn) {
      const int col = wn + tn * 16 + bn;
      bfr[tn].h[0] = *(const bf16x8*)(cB + col * LDSP + bkb);
      bfr[tn].h[1] = *(const bf16x8*)(cB + col * LDSP + bkb + 16);
    }
#pragma unroll
    for (int tm = 0; tm < 2; ++tm)
#pragma unroll
      for (int tn = 0; tn < 4; ++tn)
        acc[tm][tn] = __builtin_amdgcn_wmma_f32_16x16x32_bf16(
            false, afr[tm].v, false, bfr[tn].v,
            (short)0, acc[tm][tn], false, false);
  };

  // Prologue: stage tile 0 into buffer 0.
  async_copy_b128(pA0, &lA[0][lo]);
  async_copy_b128(pA1, &lA[0][lo + 64 * LDSP]);
  async_copy_b128(pB0, &lB[0][lo]);
  async_copy_b128(pB1, &lB[0][lo + 64 * LDSP]);
  pA0 += BK; pA1 += BK; pB0 += BK; pB1 += BK;

  // Steady state: branch-free issue-next -> wait -> barrier -> compute.
  int buf = 0;
  for (int k0 = 0; k0 < K - BK; k0 += BK) {
    const int nb = buf ^ 1;
    async_copy_b128(pA0, &lA[nb][lo]);
    async_copy_b128(pA1, &lA[nb][lo + 64 * LDSP]);
    async_copy_b128(pB0, &lB[nb][lo]);
    async_copy_b128(pB1, &lB[nb][lo + 64 * LDSP]);
    pA0 += BK; pA1 += BK; pB0 += BK; pB1 += BK;

    wait_async_le4();   // current tile's 4 copies complete (in-order per wave)
    __syncthreads();    // current tile visible to all waves
    compute_stage(lA[buf], lB[buf]);
    __syncthreads();    // all waves done reading buf before it is refilled
    buf = nb;
  }

  // Peeled last stage.
  wait_async_0();
  __syncthreads();
  compute_stage(lA[buf], lB[buf]);

  // ---- epilogue: C/D layout — lane<16: VGPR j -> M=j; lane>=16: M=8+j ----
  const int rofs = (lane >> 4) * 8;
  const int cn   = lane & 15;
#pragma unroll
  for (int tm = 0; tm < 2; ++tm) {
#pragma unroll
    for (int tn = 0; tn < 4; ++tn) {
#pragma unroll
      for (int j = 0; j < 8; ++j) {
        const int r = m0 + wm + tm * 16 + j + rofs;
        const int c = n0 + wn + tn * 16 + cn;
        float val = acc[tm][tn][j];
        if (ACT == ACT_SIGMOID) val = 1.0f / (1.0f + __expf(-val));
        if (ACT == ACT_RELUSQ)  { val = fmaxf(val, 0.f); val = val * val; }
        if (gatemul)  val *= gatemul[(size_t)r * N + c];
        if (residual) val += residual[(size_t)r * N + c];
        Cmat[(size_t)r * N + c] = (OutT)val;
      }
    }
  }
}

// ---------------------------------------------------------------------------
// Host-side orchestration.
// ---------------------------------------------------------------------------
extern "C" void kernel_launch(void* const* d_in, const int* in_sizes, int n_in,
                              void* d_out, int out_size, void* d_ws, size_t ws_size,
                              hipStream_t stream) {
  (void)in_sizes; (void)n_in; (void)out_size; (void)ws_size;

  const float* x          = (const float*)d_in[0];
  const float* ln1_w      = (const float*)d_in[1];
  const float* ln1_b      = (const float*)d_in[2];
  const float* tm_k       = (const float*)d_in[3];
  const float* tm_v       = (const float*)d_in[4];
  const float* tm_r       = (const float*)d_in[5];
  const float* Wk         = (const float*)d_in[6];
  const float* Wv         = (const float*)d_in[7];
  const float* Wr         = (const float*)d_in[8];
  const float* time_first = (const float*)d_in[9];
  const float* time_decay = (const float*)d_in[10];
  const float* Wo         = (const float*)d_in[11];
  const float* ln2_w      = (const float*)d_in[12];
  const float* ln2_b      = (const float*)d_in[13];
  const float* cm_k       = (const float*)d_in[14];
  const float* cm_r       = (const float*)d_in[15];
  const float* Wk_ffn     = (const float*)d_in[16];
  const float* Wv_ffn     = (const float*)d_in[17];
  const float* Wr_ffn     = (const float*)d_in[18];
  float* out = (float*)d_out;

  const size_t MC = (size_t)BT_SZ * C_SZ;     // 16M elements
  const size_t MF = (size_t)BT_SZ * FFN_SZ;   // 64M elements

  // Workspace layout (256B aligned regions).
  char* wp = (char*)d_ws;
  auto carve = [&](size_t bytes) {
    void* p = (void*)wp;
    wp += (bytes + 255) & ~(size_t)255;
    return p;
  };
  float*  xn   = (float*)carve(MC * 4);        // xn, later yn
  __bf16* A0   = (__bf16*)carve(MC * 2);       // kx, later bf16(r*wkv)
  __bf16* A1   = (__bf16*)carve(MC * 2);       // vx, later kx2
  __bf16* A2   = (__bf16*)carve(MC * 2);       // rx, later rx2
  float*  F0   = (float*)carve(MC * 4);        // k, later r2
  float*  F1   = (float*)carve(MC * 4);        // v
  float*  F2   = (float*)carve(MC * 4);        // r
  float*  F4   = (float*)carve(MC * 4);        // att_out
  __bf16* G0   = (__bf16*)carve(MF * 2);       // vv (relu^2)
  __bf16* tWk  = (__bf16*)carve((size_t)C_SZ * C_SZ * 2);
  __bf16* tWv  = (__bf16*)carve((size_t)C_SZ * C_SZ * 2);
  __bf16* tWr  = (__bf16*)carve((size_t)C_SZ * C_SZ * 2);
  __bf16* tWo  = (__bf16*)carve((size_t)C_SZ * C_SZ * 2);
  __bf16* tWrf = (__bf16*)carve((size_t)C_SZ * C_SZ * 2);
  __bf16* tWkf = (__bf16*)carve((size_t)C_SZ * FFN_SZ * 2);   // [FFN, C]
  __bf16* tWvf = (__bf16*)carve((size_t)FFN_SZ * C_SZ * 2);   // [C, FFN]

  const dim3 tb(32, 8);
  // in[K,N] -> out[N,K], bf16
  transpose_to_bf16<<<dim3(C_SZ / 32, C_SZ / 32), tb, 0, stream>>>(Wk, tWk, C_SZ, C_SZ);
  transpose_to_bf16<<<dim3(C_SZ / 32, C_SZ / 32), tb, 0, stream>>>(Wv, tWv, C_SZ, C_SZ);
  transpose_to_bf16<<<dim3(C_SZ / 32, C_SZ / 32), tb, 0, stream>>>(Wr, tWr, C_SZ, C_SZ);
  transpose_to_bf16<<<dim3(C_SZ / 32, C_SZ / 32), tb, 0, stream>>>(Wo, tWo, C_SZ, C_SZ);
  transpose_to_bf16<<<dim3(C_SZ / 32, C_SZ / 32), tb, 0, stream>>>(Wr_ffn, tWrf, C_SZ, C_SZ);
  transpose_to_bf16<<<dim3(FFN_SZ / 32, C_SZ / 32), tb, 0, stream>>>(Wk_ffn, tWkf, C_SZ, FFN_SZ);
  transpose_to_bf16<<<dim3(C_SZ / 32, FFN_SZ / 32), tb, 0, stream>>>(Wv_ffn, tWvf, FFN_SZ, C_SZ);

  const int ew_blocks = (int)(MC / 256);

  // ---- TimeMixFirst ----
  ln_kernel<<<BT_SZ, 256, 0, stream>>>(x, ln1_w, ln1_b, xn);
  mix3_kernel<<<ew_blocks, 256, 0, stream>>>(xn, tm_k, tm_v, tm_r, A0, A1, A2);

  const dim3 gC(C_SZ / 128, BT_SZ / 128);     // N=1024 GEMMs
  gemm_bf16_wmma<ACT_NONE, float><<<gC, 256, 0, stream>>>(A0, tWk, F0, nullptr, nullptr, BT_SZ, C_SZ, C_SZ);
  gemm_bf16_wmma<ACT_NONE, float><<<gC, 256, 0, stream>>>(A1, tWv, F1, nullptr, nullptr, BT_SZ, C_SZ, C_SZ);
  gemm_bf16_wmma<ACT_SIGMOID, float><<<gC, 256, 0, stream>>>(A2, tWr, F2, nullptr, nullptr, BT_SZ, C_SZ, C_SZ);

  // ---- TimeMixCell: WKV recurrence fused with r-gate -> bf16 ----
  wkv_gate_kernel<<<(B_SZ * C_SZ) / 256, 256, 0, stream>>>(F0, F1, F2, time_first, time_decay, A0);

  // ---- TimeMixLast: att_out = x + (r*wkv) @ Wo ----
  gemm_bf16_wmma<ACT_NONE, float><<<gC, 256, 0, stream>>>(A0, tWo, F4, x, nullptr, BT_SZ, C_SZ, C_SZ);

  // ---- ChannelMix ----
  ln_kernel<<<BT_SZ, 256, 0, stream>>>(F4, ln2_w, ln2_b, xn);
  mix2_kernel<<<ew_blocks, 256, 0, stream>>>(xn, cm_k, cm_r, A1, A2);

  const dim3 gF(FFN_SZ / 128, BT_SZ / 128);
  gemm_bf16_wmma<ACT_RELUSQ, __bf16><<<gF, 256, 0, stream>>>(A1, tWkf, G0, nullptr, nullptr, BT_SZ, FFN_SZ, C_SZ);
  gemm_bf16_wmma<ACT_SIGMOID, float><<<gC, 256, 0, stream>>>(A2, tWrf, F0, nullptr, nullptr, BT_SZ, C_SZ, C_SZ);
  // out = att_out + r2 * (vv @ Wv_ffn)  — fused into the final GEMM epilogue.
  gemm_bf16_wmma<ACT_NONE, float><<<gC, 256, 0, stream>>>(G0, tWvf, out, F4, F0, BT_SZ, C_SZ, FFN_SZ);
}
